// BiLSTM_CRF_39608188404426
// MI455X (gfx1250) — matrix operand order repaired and checked by reading, not compiled, not run on hardware
//
#include <hip/hip_runtime.h>
#include <hip/hip_bf16.h>
#include <stdint.h>

// BiLSTM-CRF for MI455X (gfx1250), wave32 + WMMA bf16.
//   T=4096, VOCAB=256, EMB=512, H=512/dir, gates=2048/dir.
// Pipeline:
//   1) fp32->bf16 weight converts + embedding gather (bf16)
//   2) pre[t] = x[t] @ W_ih^T + b_ih + b_hh   (WMMA GEMM, both dirs)
//   3) recurrence: 8 WGs per direction, W_hh slice (256KB bf16) resident in
//      LDS in WMMA A-fragment layout; per step: WMMA matvec + local LSTM cell,
//      h broadcast through L2, monotonic atomic-counter barrier.
//   4) feats = [hf,hb] @ W_tag^T + b_tag      (WMMA GEMM)
//   5) Viterbi with transitions resident in LDS + backtrack.

#define T_LEN 4096
#define EMB   512
#define HID   1024
#define HHALF 512
#define G4    2048
#define NEGV  (-10000.0f)

typedef __attribute__((ext_vector_type(16))) __bf16 v16bf;
typedef __attribute__((ext_vector_type(8)))  float  v8f;

union FragB16 {
  v16bf v;
  unsigned int u[8];
  uint4 q[2];
};

__device__ __forceinline__ unsigned short f2bf(float f) {
  unsigned int x = __builtin_bit_cast(unsigned int, f);
  x += 0x7fffu + ((x >> 16) & 1u);          // round-to-nearest-even
  return (unsigned short)(x >> 16);
}

__device__ __forceinline__ float sigm(float x) {
  return 1.0f / (1.0f + __expf(-x));
}

// ---------------------------------------------------------------- converts
__global__ void f32_to_bf16_kernel(const float* __restrict__ in,
                                   unsigned short* __restrict__ out, int n) {
  for (int i = blockIdx.x * blockDim.x + threadIdx.x; i < n;
       i += gridDim.x * blockDim.x)
    out[i] = f2bf(in[i]);
}

__global__ void embed_bf16_kernel(const int* __restrict__ src,
                                  const float* __restrict__ emb,
                                  unsigned short* __restrict__ xbf) {
  int idx = blockIdx.x * blockDim.x + threadIdx.x;
  if (idx < T_LEN * EMB) {
    int t = idx >> 9, e = idx & 511;
    xbf[idx] = f2bf(emb[src[t] * EMB + e]);
  }
}

// ---------------------------------------------------------- WMMA GEMM
// C[M][N] = A(MxK,bf16) @ W(NxK,bf16)^T + bias1 + bias2   (fp32 out)
// One wave per 16x16 C tile. M,N mult of 16; K mult of 32.
__global__ void __launch_bounds__(256) gemm_bf16_kernel(
    const unsigned short* __restrict__ A, const unsigned short* __restrict__ W,
    const float* __restrict__ bias1, const float* __restrict__ bias2,
    float* __restrict__ C, int M, int N, int K) {
  const int lane = threadIdx.x;
  const int tile = blockIdx.x * blockDim.y + threadIdx.y;
  const int ntN = N >> 4;
  const int mt = tile / ntN, nt = tile - mt * ntN;
  const int l15 = lane & 15;
  const int hi = (lane & 16) ? 1 : 0;
  const unsigned short* arow = A + (size_t)(mt * 16 + l15) * K;
  const unsigned short* brow = W + (size_t)(nt * 16 + l15) * K;

  v8f acc = {};
  const int nk = K >> 5;
  for (int kc = 0; kc < nk; ++kc) {
    FragB16 a, b;
    // A 16x32 bf16 fragment: lanes0-15 K{0..7,16..23}, lanes16-31 K{8..15,24..31}
    int ka = kc * 32 + hi * 8;
    a.q[0] = *(const uint4*)(arow + ka);
    a.q[1] = *(const uint4*)(arow + ka + 16);
    // B 32x16 bf16 fragment: lanes0-15 K{0..7,16..23} @ N=lane, lanes16-31 K{8..15,24..31}
    int kb = kc * 32 + hi * 8;
    b.q[0] = *(const uint4*)(brow + kb);
    b.q[1] = *(const uint4*)(brow + kb + 16);
    acc = __builtin_amdgcn_wmma_f32_16x16x32_bf16(false, a.v, false, b.v,
                                                  (short)0, acc, false, false);
  }
  const int n = nt * 16 + l15;
  float bb = bias1 ? bias1[n] : 0.0f;
  if (bias2) bb += bias2[n];
  const int mb = mt * 16 + hi * 8;
#pragma unroll
  for (int r = 0; r < 8; ++r)
    C[(size_t)(mb + r) * N + n] = acc[r] + bb;
}

// ---------------------------------------------------------- recurrence
// 16 blocks x 256 threads: block b: dir = b>>3, slice r = b&7.
// Slice owns hidden units u in [64r,64r+64) and gate rows {512q+64r+u}.
// W_hh slice pre-packed into LDS as 256 WMMA A-fragments (256KB).
__global__ void __launch_bounds__(256, 1) lstm_bidir_kernel(
    const unsigned short* __restrict__ WhhF,
    const unsigned short* __restrict__ WhhB,
    const float* __restrict__ preF, const float* __restrict__ preB,
    float* __restrict__ hs, unsigned int* __restrict__ counters) {
  extern __shared__ char smem[];
  unsigned int* wfrag = (unsigned int*)smem;      // 65536 dwords = 256KB
  float* h_s   = (float*)(smem + 262144);         // 512 fp32
  float* pre_s = h_s + 512;                       // 256 fp32
  float* gates = pre_s + 256;                     // 256 fp32

  const int tid  = threadIdx.x;
  const int lane = tid & 31;
  const int wv   = tid >> 5;                      // 8 waves
  const int dir  = blockIdx.x >> 3;
  const int r    = blockIdx.x & 7;
  const unsigned short* Whh = dir ? WhhB : WhhF;
  const float* pre = dir ? preB : preF;

  // Prologue: gather this slice's 256 W_hh rows into A-fragment layout.
  for (int d = tid; d < 65536; d += 256) {
    int frag = d >> 8;                 // (mtile, kchunk)
    int w = d & 255;
    int fl = w >> 3;                   // fragment lane
    int j = w & 7;                     // fragment vgpr (dword)
    int mt = frag >> 4;
    int kc = frag & 15;
    int lr = mt * 16 + (fl & 15);      // local row 0..255
    int gr = (lr >> 6) * HHALF + 64 * r + (lr & 63);   // global gate row
    int base0 = kc * 32 + ((fl & 16) ? 16 : 0);
    int k = base0 + 2 * j;
    unsigned int lo = Whh[(size_t)gr * HHALF + k];
    unsigned int hh = Whh[(size_t)gr * HHALF + k + 1];
    wfrag[d] = lo | (hh << 16);
  }
  __syncthreads();

  float creg = 0.0f;  // cell state for unit tid (threads 0..63)
  const int gr_tid = (tid >> 6) * HHALF + 64 * r + (tid & 63);

  for (int s = 0; s < T_LEN; ++s) {
    const int t = dir ? (T_LEN - 1 - s) : s;
    if (s == 0) {
      h_s[tid] = 0.0f; h_s[tid + 256] = 0.0f;
    } else {
      const int tp = dir ? (t + 1) : (t - 1);
      const float* hp = hs + (size_t)tp * HID + dir * HHALF;
      h_s[tid] = hp[tid]; h_s[tid + 256] = hp[tid + 256];
    }
    pre_s[tid] = pre[(size_t)t * G4 + gr_tid];
    if (s + 1 < T_LEN) {
      const int tn = dir ? (t - 1) : (t + 1);
      __builtin_prefetch(&pre[(size_t)tn * G4 + gr_tid], 0, 0);
    }
    __syncthreads();

    // gates(256x1) += Wslice(256x512) @ h(512x1): wave wv owns tiles 2wv,2wv+1
    v8f acc0 = {}, acc1 = {};
    const int hi = (lane & 16) ? 1 : 0;
    const bool bact = ((lane & 15) == 0);   // lanes holding B column N=0
#pragma unroll 4
    for (int kc = 0; kc < 16; ++kc) {
      FragB16 b;
      int kb = kc * 32 + hi * 16;
#pragma unroll
      for (int j = 0; j < 8; ++j) {
        float x0 = bact ? h_s[kb + 2 * j] : 0.0f;
        float x1 = bact ? h_s[kb + 2 * j + 1] : 0.0f;
        b.u[j] = (unsigned int)f2bf(x0) | ((unsigned int)f2bf(x1) << 16);
      }
      FragB16 a0, a1;
      const unsigned int* f0 = wfrag + ((((2 * wv) * 16 + kc) << 8) + lane * 8);
      const unsigned int* f1 = wfrag + ((((2 * wv + 1) * 16 + kc) << 8) + lane * 8);
      a0.q[0] = *(const uint4*)(f0); a0.q[1] = *(const uint4*)(f0 + 4);
      a1.q[0] = *(const uint4*)(f1); a1.q[1] = *(const uint4*)(f1 + 4);
      acc0 = __builtin_amdgcn_wmma_f32_16x16x32_bf16(false, a0.v, false, b.v,
                                                     (short)0, acc0, false, false);
      acc1 = __builtin_amdgcn_wmma_f32_16x16x32_bf16(false, a1.v, false, b.v,
                                                     (short)0, acc1, false, false);
    }
    if (bact) {  // D column 0 lives in lanes 0 (M0..7) and 16 (M8..15)
      int mb = hi * 8;
#pragma unroll
      for (int rr = 0; rr < 8; ++rr) {
        int lr0 = (2 * wv) * 16 + mb + rr;
        int lr1 = (2 * wv + 1) * 16 + mb + rr;
        gates[lr0] = acc0[rr] + pre_s[lr0];
        gates[lr1] = acc1[rr] + pre_s[lr1];
      }
    }
    __syncthreads();

    if (tid < 64) {  // LSTM cell for local hidden units
      float gi = gates[tid], gf = gates[64 + tid];
      float gg = gates[128 + tid], go = gates[192 + tid];
      creg = sigm(gf) * creg + sigm(gi) * tanhf(gg);
      float h = sigm(go) * tanhf(creg);
      hs[(size_t)t * HID + dir * HHALF + 64 * r + tid] = h;
    }
    __threadfence();
    if (tid == 0) {  // monotonic counter barrier across the 8 slices of dir
      atomicAdd(&counters[dir], 1u);
      while (__hip_atomic_load(&counters[dir], __ATOMIC_ACQUIRE,
                               __HIP_MEMORY_SCOPE_AGENT) <
             8u * (unsigned)(s + 1)) {
        __builtin_amdgcn_s_sleep(1);
      }
    }
    __syncthreads();
    __threadfence();
  }
}

// ---------------------------------------------------------- Viterbi + backtrack
__global__ void __launch_bounds__(1024, 1) viterbi_kernel(
    const float* __restrict__ feats, const float* __restrict__ trans,
    float* __restrict__ out, unsigned char* __restrict__ bp) {
  extern __shared__ char smem[];
  float* trans_s = (float*)smem;        // 65536 fp32 = 256KB
  float* fv     = trans_s + 65536;
  float* nfv    = fv + 256;
  float* feat_s = nfv + 256;
  const int tid = threadIdx.x;

  for (int i = tid; i < 65536; i += 1024) trans_s[i] = trans[i];
  if (tid < 256) fv[tid] = (tid == 0) ? 0.0f : NEGV;  // START tag == 0
  __syncthreads();

  const int i = tid >> 2;     // tag, 4 lanes each
  const int sub = tid & 3;
  const int j0 = sub * 64;
  for (int t = 0; t < T_LEN; ++t) {
    if (tid < 256) feat_s[tid] = feats[t * 256 + tid];
    __syncthreads();
    float m = fv[j0] + trans_s[i * 256 + j0];
    int am = j0;
#pragma unroll 8
    for (int jj = 1; jj < 64; ++jj) {
      int j = j0 + jj;
      float v = fv[j] + trans_s[i * 256 + j];
      if (v > m) { m = v; am = j; }
    }
#pragma unroll
    for (int off = 1; off < 4; off <<= 1) {   // reduce over sub-lanes
      float om = __shfl_xor(m, off, 32);
      int   oa = __shfl_xor(am, off, 32);
      if (om > m || (om == m && oa < am)) { m = om; am = oa; }
    }
    if (sub == 0) {
      nfv[i] = m + feat_s[i];
      bp[t * 256 + i] = (unsigned char)am;
    }
    __syncthreads();
    if (tid < 256) fv[tid] = nfv[tid];
    __syncthreads();
  }

  if (tid < 256) nfv[tid] = fv[tid] + trans_s[1 * 256 + tid];  // STOP row == 1
  __threadfence_block();
  __syncthreads();
  if (tid == 0) {
    int best = 0; float bsc = nfv[0];
    for (int k = 1; k < 256; ++k)
      if (nfv[k] > bsc) { bsc = nfv[k]; best = k; }
    out[0] = bsc;
    int tag = best;
    for (int t = T_LEN - 1; t >= 0; --t) {
      out[1 + t] = (float)tag;
      tag = bp[t * 256 + tag];
    }
  }
}

// ---------------------------------------------------------------- launch
extern "C" void kernel_launch(void* const* d_in, const int* in_sizes, int n_in,
                              void* d_out, int out_size, void* d_ws,
                              size_t ws_size, hipStream_t stream) {
  (void)in_sizes; (void)n_in; (void)out_size; (void)ws_size;
  const int*   src    = (const int*)d_in[0];
  const float* emb    = (const float*)d_in[2];
  const float* W_ih_f = (const float*)d_in[3];
  const float* W_hh_f = (const float*)d_in[4];
  const float* b_ih_f = (const float*)d_in[5];
  const float* b_hh_f = (const float*)d_in[6];
  const float* W_ih_b = (const float*)d_in[7];
  const float* W_hh_b = (const float*)d_in[8];
  const float* b_ih_b = (const float*)d_in[9];
  const float* b_hh_b = (const float*)d_in[10];
  const float* W_tag  = (const float*)d_in[11];
  const float* b_tag  = (const float*)d_in[12];
  const float* trans  = (const float*)d_in[13];

  char* ws = (char*)d_ws;
  size_t off = 0;
  auto alloc = [&](size_t bytes) -> char* {
    char* p = ws + off;
    off = (off + bytes + 255) & ~(size_t)255;
    return p;
  };
  unsigned short* xbf     = (unsigned short*)alloc((size_t)T_LEN * EMB * 2);
  unsigned short* wihf_bf = (unsigned short*)alloc((size_t)G4 * EMB * 2);
  unsigned short* whhf_bf = (unsigned short*)alloc((size_t)G4 * HHALF * 2);
  unsigned short* wihb_bf = (unsigned short*)alloc((size_t)G4 * EMB * 2);
  unsigned short* whhb_bf = (unsigned short*)alloc((size_t)G4 * HHALF * 2);
  unsigned short* wtag_bf = (unsigned short*)alloc((size_t)256 * HID * 2);
  float* pre_f            = (float*)alloc((size_t)T_LEN * G4 * 4);
  float* pre_b            = (float*)alloc((size_t)T_LEN * G4 * 4);
  float* hsbuf            = (float*)alloc((size_t)T_LEN * HID * 4);
  unsigned short* hs_bf   = (unsigned short*)alloc((size_t)T_LEN * HID * 2);
  float* feats            = (float*)alloc((size_t)T_LEN * 256 * 4);
  unsigned char* bp       = (unsigned char*)alloc((size_t)T_LEN * 256);
  unsigned int* counters  = (unsigned int*)alloc(256);

  const size_t LSTM_LDS = 262144 + (512 + 256 + 256) * sizeof(float);
  const size_t VIT_LDS  = (65536 + 768) * sizeof(float);
  hipFuncSetAttribute((const void*)lstm_bidir_kernel,
                      hipFuncAttributeMaxDynamicSharedMemorySize, (int)LSTM_LDS);
  hipFuncSetAttribute((const void*)viterbi_kernel,
                      hipFuncAttributeMaxDynamicSharedMemorySize, (int)VIT_LDS);

  // 1) weight converts + embedding gather
  f32_to_bf16_kernel<<<2048, 256, 0, stream>>>(W_ih_f, wihf_bf, G4 * EMB);
  f32_to_bf16_kernel<<<2048, 256, 0, stream>>>(W_hh_f, whhf_bf, G4 * HHALF);
  f32_to_bf16_kernel<<<2048, 256, 0, stream>>>(W_ih_b, wihb_bf, G4 * EMB);
  f32_to_bf16_kernel<<<2048, 256, 0, stream>>>(W_hh_b, whhb_bf, G4 * HHALF);
  f32_to_bf16_kernel<<<1024, 256, 0, stream>>>(W_tag, wtag_bf, 256 * HID);
  embed_bf16_kernel<<<(T_LEN * EMB + 255) / 256, 256, 0, stream>>>(src, emb, xbf);

  // 2) input-gate GEMMs: (4096x512)@(512x2048) per direction
  {
    dim3 blk(32, 8);
    int tiles = (T_LEN / 16) * (G4 / 16);   // 32768 waves
    gemm_bf16_kernel<<<tiles / 8, blk, 0, stream>>>(
        xbf, wihf_bf, b_ih_f, b_hh_f, pre_f, T_LEN, G4, EMB);
    gemm_bf16_kernel<<<tiles / 8, blk, 0, stream>>>(
        xbf, wihb_bf, b_ih_b, b_hh_b, pre_b, T_LEN, G4, EMB);
  }

  // 3) recurrence: 8 WGs/direction, LDS-resident W_hh slices
  hipMemsetAsync(counters, 0, 256, stream);
  lstm_bidir_kernel<<<16, 256, LSTM_LDS, stream>>>(whhf_bf, whhb_bf, pre_f,
                                                   pre_b, hsbuf, counters);

  // 4) tag projection: (4096x1024)@(1024x256)
  f32_to_bf16_kernel<<<4096, 256, 0, stream>>>(hsbuf, hs_bf, T_LEN * HID);
  {
    dim3 blk(32, 8);
    int tiles = (T_LEN / 16) * (256 / 16);  // 4096 waves
    gemm_bf16_kernel<<<tiles / 8, blk, 0, stream>>>(
        hs_bf, wtag_bf, b_tag, (const float*)nullptr, feats, T_LEN, 256, HID);
  }

  // 5) Viterbi + backtrack
  viterbi_kernel<<<1, 1024, VIT_LDS, stream>>>(feats, trans, (float*)d_out, bp);
}
